// MultiHeadSelfAttn_25640954757735
// MI455X (gfx1250) — matrix operand order repaired and checked by reading, not compiled, not run on hardware
//
#include <hip/hip_runtime.h>
#include <hip/hip_bf16.h>

// ---------------------------------------------------------------------------
// Multi-head self-attention forward for MI455X (gfx1250), bf16 WMMA pipeline.
// B=8, T=1024, C=512, H=8, HD=64.  d_out = [out (B*T*C) | attn (B*H*T*T)] f32.
//
// - All WMMA B-operands are K-contiguous (transposed weights, Vt): one 32B
//   load per fragment per lane.
// - GEMMs register-block 32x64 per wave: 8 WMMAs per 6 fragment loads.
// - Attention blocks cover 32 queries; K fragments reused across 2 q-subtiles.
// - attn / out results are streamed with non-temporal stores (never re-read),
//   preserving L2 for K/V/weights.
// ---------------------------------------------------------------------------

constexpr int B_  = 8;
constexpr int T_  = 1024;
constexpr int C_  = 512;
constexpr int H_  = 8;
constexpr int HD_ = 64;
constexpr float FMIN_ = -3.402823466e38f;

typedef __attribute__((ext_vector_type(16))) __bf16 v16bf;
typedef __attribute__((ext_vector_type(8)))  __bf16 v8bf;
typedef __attribute__((ext_vector_type(8)))  float  v8f;

// ---- WMMA: D = A(16x32 bf16) * B(32x16 bf16) + C(16x16 f32) ---------------
__device__ __forceinline__ v8f wmma_bf16(v16bf a, v16bf b, v8f c) {
    return __builtin_amdgcn_wmma_f32_16x16x32_bf16(
        false, a, false, b, (short)0, c, false, false);
}

// A fragment, row-major source. Lane l holds row (l&15); lanes<16 carry
// K {0..15}, lanes>=16 carry {16..31} (matches frag_b_contig).
__device__ __forceinline__ v16bf frag_a_row(const __bf16* __restrict__ row,
                                            int k0, int lane) {
    const int kb = (lane >> 4) << 4;          // 0 or 16
    v8bf lo = *(const v8bf*)(row + k0 + kb);
    v8bf hi = *(const v8bf*)(row + k0 + kb + 8);
    v16bf a;
#pragma unroll
    for (int j = 0; j < 8; ++j) { a[j] = lo[j]; a[8 + j] = hi[j]; }
    return a;
}

// B fragment when K is contiguous for this column (col = base of K-run).
// Lanes 0-15 hold K=k0..k0+15, lanes 16-31 hold K=k0+16..k0+31.
__device__ __forceinline__ v16bf frag_b_contig(const __bf16* __restrict__ col,
                                               int k0, int lane) {
    const int kh = (lane >> 4) << 4;          // 0 or 16
    return *(const v16bf*)(col + k0 + kh);    // one 32B load
}

// ---------------------------------------------------------------------------
__global__ void cvt_f32_bf16_kernel(const float* __restrict__ src,
                                    __bf16* __restrict__ dst, int n) {
    int i = blockIdx.x * blockDim.x + threadIdx.x;
    if (i < n) dst[i] = (__bf16)src[i];
}

// Convert + transpose a CxC weight: dst[n*C + k] = (bf16)src[k*C + n].
__global__ void cvt_transpose_kernel(const float* __restrict__ src,
                                     __bf16* __restrict__ dst) {
    int i = blockIdx.x * blockDim.x + threadIdx.x;   // i = n*C + k
    if (i < C_ * C_) {
        int n = i >> 9, k = i & 511;
        dst[i] = (__bf16)src[k * C_ + n];
    }
}

// ---------------------------------------------------------------------------
// QKV projection: Y = X @ W{q,k,v} + b.
// q,k scattered to [B][H][T][HD]; v scattered TRANSPOSED to [B][H][HD][T].
// grid = (B*T/32, 3*C/256), block = 128; each wave computes a 32x64 strip.
__global__ void qkv_gemm_kernel(const __bf16* __restrict__ X,
                                const __bf16* __restrict__ WtQKV,  // 3*C*C [n][k]
                                const float* __restrict__ bq,
                                const float* __restrict__ bk,
                                const float* __restrict__ bv,
                                __bf16* __restrict__ qkv) {        // 3 * B*T*C
    const int tid  = threadIdx.x;
    const int wave = tid >> 5, lane = tid & 31;
    const int mcol = lane & 15;
    const int mbase = blockIdx.x << 5;                   // 32 rows
    const int nbase = (blockIdx.y << 8) + (wave << 6);   // [0,1536), 64 cols
    const int g     = nbase >> 9;                        // 0=q 1=k 2=v
    const int ncol  = nbase & 511;
    const __bf16* Wt   = WtQKV + (size_t)g * (C_ * C_);
    const float*  bias = (g == 0) ? bq : ((g == 1) ? bk : bv);
    const __bf16* arow0 = X + (size_t)(mbase + mcol) * C_;
    const __bf16* arow1 = X + (size_t)(mbase + 16 + mcol) * C_;

    const __bf16* bp0 = Wt + (size_t)(ncol +  0 + mcol) * C_;
    const __bf16* bp1 = Wt + (size_t)(ncol + 16 + mcol) * C_;
    const __bf16* bp2 = Wt + (size_t)(ncol + 32 + mcol) * C_;
    const __bf16* bp3 = Wt + (size_t)(ncol + 48 + mcol) * C_;

    v8f acc[2][4] = {};
#pragma unroll 2
    for (int k0 = 0; k0 < C_; k0 += 32) {
        __builtin_prefetch(arow0 + k0 + 128, 0, 3);
        __builtin_prefetch(arow1 + k0 + 128, 0, 3);
        v16bf a0 = frag_a_row(arow0, k0, lane);
        v16bf a1 = frag_a_row(arow1, k0, lane);
        v16bf b0 = frag_b_contig(bp0, k0, lane);
        v16bf b1 = frag_b_contig(bp1, k0, lane);
        v16bf b2 = frag_b_contig(bp2, k0, lane);
        v16bf b3 = frag_b_contig(bp3, k0, lane);
        acc[0][0] = wmma_bf16(a0, b0, acc[0][0]);
        acc[1][0] = wmma_bf16(a1, b0, acc[1][0]);
        acc[0][1] = wmma_bf16(a0, b1, acc[0][1]);
        acc[1][1] = wmma_bf16(a1, b1, acc[1][1]);
        acc[0][2] = wmma_bf16(a0, b2, acc[0][2]);
        acc[1][2] = wmma_bf16(a1, b2, acc[1][2]);
        acc[0][3] = wmma_bf16(a0, b3, acc[0][3]);
        acc[1][3] = wmma_bf16(a1, b3, acc[1][3]);
    }

    const int Mh = (lane >> 4) << 3;
#pragma unroll
    for (int i = 0; i < 4; ++i) {
        const int n  = ncol + (i << 4) + mcol;           // col within matrix g
        const int h  = n >> 6, hd = n & 63;
        const float bn = bias[n];
#pragma unroll
        for (int s = 0; s < 2; ++s) {
#pragma unroll
            for (int r = 0; r < 8; ++r) {
                const int mg = mbase + (s << 4) + Mh + r;
                const int bb = mg >> 10, tt = mg & 1023;
                const __bf16 val = (__bf16)(acc[s][i][r] + bn);
                if (g < 2) {      // q, k: token-major [B][H][T][HD]
                    qkv[(size_t)g * (B_ * T_ * C_) +
                        (((size_t)(bb * H_ + h) * T_ + tt) << 6) + hd] = val;
                } else {          // v: transposed [B][H][HD][T]
                    qkv[(size_t)2 * (B_ * T_ * C_) +
                        (((size_t)(bb * H_ + h) * HD_ + hd) << 10) + tt] = val;
                }
            }
        }
    }
}

// ---------------------------------------------------------------------------
// Fused attention for one (b, h, 32-query tile).
// grid = (T/32, H, B), block = 256 (8 waves),
// dynamic LDS = 32*T*4 (f32 S) + 32*T*2 (bf16 P) = 192 KB (<320 KB/WGP).
__global__ void attn_kernel(const __bf16* __restrict__ q,   // [B][H][T][HD]
                            const __bf16* __restrict__ k,   // [B][H][T][HD]
                            const __bf16* __restrict__ vt,  // [B][H][HD][T]
                            const float* __restrict__ mask,
                            float* __restrict__ attn_out,   // B*H*T*T
                            __bf16* __restrict__ attend) {  // B*T*C
    extern __shared__ char smem[];
    float*  S   = (float*)smem;                              // [32][T_]
    __bf16* Pbf = (__bf16*)(smem + 32 * T_ * sizeof(float)); // [32][T_]

    const int qt = blockIdx.x, h = blockIdx.y, b = blockIdx.z;
    const int tid = threadIdx.x, wave = tid >> 5, lane = tid & 31;
    const size_t headoff = (size_t)((b * H_ + h) * T_) * HD_;
    const __bf16* qh  = q  + headoff;
    const __bf16* kh  = k  + headoff;
    const __bf16* vth = vt + headoff;                        // [HD][T]
    const int mcol = lane & 15;
    const int Mh   = (lane >> 4) << 3;

    // -------- Phase 1: S = scale * (q_tile @ K^T), masked ------------------
    // Each wave owns key-tiles kt = wave + 8i and computes BOTH q-subtiles,
    // reusing the K B-fragments (4 WMMAs per 2 B loads).
    {
        const __bf16* ar0 = qh + (size_t)(qt * 32 + mcol) * HD_;
        const __bf16* ar1 = qh + (size_t)(qt * 32 + 16 + mcol) * HD_;
        const v16bf a00 = frag_a_row(ar0, 0, lane);   // loop-invariant q frags
        const v16bf a01 = frag_a_row(ar0, 32, lane);
        const v16bf a10 = frag_a_row(ar1, 0, lane);
        const v16bf a11 = frag_a_row(ar1, 32, lane);
        for (int kt = wave; kt < T_ / 16; kt += 8) {
            const int key = (kt << 4) + mcol;
            const __bf16* bcol = kh + (size_t)key * HD_;
            const v16bf b0 = frag_b_contig(bcol, 0, lane);
            const v16bf b1 = frag_b_contig(bcol, 32, lane);
            v8f acc0 = {}, acc1 = {};
            acc0 = wmma_bf16(a00, b0, acc0);
            acc1 = wmma_bf16(a10, b0, acc1);
            acc0 = wmma_bf16(a01, b1, acc0);
            acc1 = wmma_bf16(a11, b1, acc1);
            const float mv  = mask[b * T_ + key];
            const float neg = (1.0f - mv) * FMIN_;
#pragma unroll
            for (int r = 0; r < 8; ++r) {
                S[(Mh + r) * T_ + key]        = (acc0[r] * 0.125f) * mv + neg;
                S[(16 + Mh + r) * T_ + key]   = (acc1[r] * 0.125f) * mv + neg;
            }
        }
    }
    __syncthreads();

    // -------- Phase 2: row softmax; emit fp32 attn (NT) + bf16 P (LDS) -----
    {
        const int row = tid >> 3, seg = tid & 7;      // 32 rows x 8 lanes
        float* Srow = S + row * T_;
        __bf16* Prow = Pbf + row * T_;
        const int c0 = seg << 7;                      // 128 cols per thread
        float lmax = -3.402823466e38f;
        for (int c = 0; c < 128; ++c) lmax = fmaxf(lmax, Srow[c0 + c]);
#pragma unroll
        for (int off = 1; off < 8; off <<= 1)
            lmax = fmaxf(lmax, __shfl_xor(lmax, off, 8));
        float psum = 0.0f;
        for (int c = 0; c < 128; ++c) {
            const float e = __expf(Srow[c0 + c] - lmax);
            Srow[c0 + c] = e;
            psum += e;
        }
#pragma unroll
        for (int off = 1; off < 8; off <<= 1)
            psum += __shfl_xor(psum, off, 8);
        const float inv = 1.0f / psum;
        const int   qg  = (qt << 5) + row;
        float* aout = attn_out + ((size_t)((b * H_ + h) * T_) + qg) * T_;
        for (int c = 0; c < 128; ++c) {
            const float p = Srow[c0 + c] * inv;
            __builtin_nontemporal_store(p, aout + c0 + c); // stream, bypass L2
            Prow[c0 + c] = (__bf16)p;                      // bf16 P for P@V
        }
    }
    __syncthreads();

    // -------- Phase 3: attend = P @ V (8 waves x one 16x16 tile) -----------
    {
        const int qsub = wave >> 2;                   // 0/1
        const int n    = ((wave & 3) << 4) + mcol;    // hd column
        const __bf16* prow = Pbf + (size_t)((qsub << 4) + mcol) * T_;
        const __bf16* bcol = vth + (size_t)n * T_;    // K(=token)-contiguous
        v8f acc = {};
#pragma unroll 4
        for (int kt = 0; kt < T_; kt += 32) {
            v16bf a = frag_a_row(prow, kt, lane);     // two ds_load_b128
            acc = wmma_bf16(a, frag_b_contig(bcol, kt, lane), acc);
        }
#pragma unroll
        for (int r = 0; r < 8; ++r) {
            const int qg = (qt << 5) + (qsub << 4) + Mh + r;
            attend[(size_t)(b * T_ + qg) * C_ + h * HD_ + n] = (__bf16)acc[r];
        }
    }
}

// ---------------------------------------------------------------------------
// Output projection: out = (attend @ Wo + bo) * mask[b,t], fp32 (NT stores).
// grid = (B*T/32, C/256), block = 128; each wave computes a 32x64 strip.
__global__ void out_gemm_kernel(const __bf16* __restrict__ A,     // B*T x C
                                const __bf16* __restrict__ WoT,   // [n][k]
                                const float* __restrict__ bo,
                                const float* __restrict__ mask,
                                float* __restrict__ out) {
    const int tid  = threadIdx.x;
    const int wave = tid >> 5, lane = tid & 31;
    const int mcol = lane & 15;
    const int mbase = blockIdx.x << 5;
    const int ncol  = (blockIdx.y << 8) + (wave << 6);
    const __bf16* arow0 = A + (size_t)(mbase + mcol) * C_;
    const __bf16* arow1 = A + (size_t)(mbase + 16 + mcol) * C_;
    const __bf16* bp0 = WoT + (size_t)(ncol +  0 + mcol) * C_;
    const __bf16* bp1 = WoT + (size_t)(ncol + 16 + mcol) * C_;
    const __bf16* bp2 = WoT + (size_t)(ncol + 32 + mcol) * C_;
    const __bf16* bp3 = WoT + (size_t)(ncol + 48 + mcol) * C_;

    v8f acc[2][4] = {};
#pragma unroll 2
    for (int k0 = 0; k0 < C_; k0 += 32) {
        __builtin_prefetch(arow0 + k0 + 128, 0, 3);
        __builtin_prefetch(arow1 + k0 + 128, 0, 3);
        v16bf a0 = frag_a_row(arow0, k0, lane);
        v16bf a1 = frag_a_row(arow1, k0, lane);
        v16bf b0 = frag_b_contig(bp0, k0, lane);
        v16bf b1 = frag_b_contig(bp1, k0, lane);
        v16bf b2 = frag_b_contig(bp2, k0, lane);
        v16bf b3 = frag_b_contig(bp3, k0, lane);
        acc[0][0] = wmma_bf16(a0, b0, acc[0][0]);
        acc[1][0] = wmma_bf16(a1, b0, acc[1][0]);
        acc[0][1] = wmma_bf16(a0, b1, acc[0][1]);
        acc[1][1] = wmma_bf16(a1, b1, acc[1][1]);
        acc[0][2] = wmma_bf16(a0, b2, acc[0][2]);
        acc[1][2] = wmma_bf16(a1, b2, acc[1][2]);
        acc[0][3] = wmma_bf16(a0, b3, acc[0][3]);
        acc[1][3] = wmma_bf16(a1, b3, acc[1][3]);
    }

    const int Mh = (lane >> 4) << 3;
#pragma unroll
    for (int i = 0; i < 4; ++i) {
        const int n  = ncol + (i << 4) + mcol;
        const float bn = bo[n];
#pragma unroll
        for (int s = 0; s < 2; ++s) {
#pragma unroll
            for (int r = 0; r < 8; ++r) {
                const int mg = mbase + (s << 4) + Mh + r;
                const float val = (acc[s][i][r] + bn) * mask[mg];
                __builtin_nontemporal_store(val, &out[(size_t)mg * C_ + n]);
            }
        }
    }
}

// ---------------------------------------------------------------------------
extern "C" void kernel_launch(void* const* d_in, const int* in_sizes, int n_in,
                              void* d_out, int out_size, void* d_ws, size_t ws_size,
                              hipStream_t stream) {
    const float* x    = (const float*)d_in[0];
    const float* mask = (const float*)d_in[1];
    const float* Wq   = (const float*)d_in[2];
    const float* bq   = (const float*)d_in[3];
    const float* Wk   = (const float*)d_in[4];
    const float* bk   = (const float*)d_in[5];
    const float* Wv   = (const float*)d_in[6];
    const float* bv   = (const float*)d_in[7];
    const float* Wo   = (const float*)d_in[8];
    const float* bo   = (const float*)d_in[9];

    constexpr size_t NX = (size_t)B_ * T_ * C_;      // 4,194,304
    constexpr size_t NW = (size_t)C_ * C_;           // 262,144

    // Workspace (bf16): X | WqT|WkT|WvT | WoT | q|k|vt | attend  (~42 MB)
    __bf16* Xbf   = (__bf16*)d_ws;
    __bf16* WtQKV = Xbf   + NX;
    __bf16* WoT   = WtQKV + 3 * NW;
    __bf16* qkv   = WoT   + NW;
    __bf16* att   = qkv   + 3 * NX;

    float* out  = (float*)d_out;
    float* attn = out + NX;                          // B*H*T*T

    // 1) conversions (weights also transposed to [n][k])
    cvt_f32_bf16_kernel<<<(int)((NX + 255) / 256), 256, 0, stream>>>(x, Xbf, (int)NX);
    const int wblk = (int)((NW + 255) / 256);
    cvt_transpose_kernel<<<wblk, 256, 0, stream>>>(Wq, WtQKV);
    cvt_transpose_kernel<<<wblk, 256, 0, stream>>>(Wk, WtQKV + NW);
    cvt_transpose_kernel<<<wblk, 256, 0, stream>>>(Wv, WtQKV + 2 * NW);
    cvt_transpose_kernel<<<wblk, 256, 0, stream>>>(Wo, WoT);

    // 2) QKV projection (v written transposed per head)
    dim3 g1(B_ * T_ / 32, 3 * C_ / 256);
    qkv_gemm_kernel<<<g1, 128, 0, stream>>>(Xbf, WtQKV, bq, bk, bv, qkv);

    // 3) fused attention (scores + softmax + attn write + P@V)
    dim3 g2(T_ / 32, H_, B_);
    const size_t smem = (size_t)32 * T_ * (sizeof(float) + sizeof(__bf16)); // 192 KB
    attn_kernel<<<g2, 256, smem, stream>>>(qkv, qkv + NX, qkv + 2 * NX,
                                           mask, attn, att);

    // 4) output projection + mask
    dim3 g3(B_ * T_ / 32, C_ / 256);
    out_gemm_kernel<<<g3, 128, 0, stream>>>(att, WoT, bo, mask, out);
}
